// Attention_87926570484382
// MI455X (gfx1250) — matrix operand order repaired and checked by reading, not compiled, not run on hardware
//
#include <hip/hip_runtime.h>

#define T_DIM  128
#define SIN_D  64
#define EMBED  512
#define HEADS  8
#define HDIM   64
#define BATCH  16
#define SEQ    1024
#define QKVC   1536
#define MROWS  (BATCH*SEQ)

typedef __attribute__((ext_vector_type(4)))  float    v4f;
typedef __attribute__((ext_vector_type(8)))  float    v8f;
typedef __attribute__((ext_vector_type(8)))  _Float16 v8h;
typedef __attribute__((ext_vector_type(16))) _Float16 v16h;

static __device__ __forceinline__ v16h cat16(v8h lo, v8h hi) {
  return __builtin_shufflevector(lo, hi, 0,1,2,3,4,5,6,7,8,9,10,11,12,13,14,15);
}
static __device__ __forceinline__ v8h ld8(const _Float16* p) { return *(const v8h*)p; }

#define WMMA(A,B,C) __builtin_amdgcn_wmma_f32_16x16x32_f16(false,(A),false,(B),(short)0,(C),false,false)

// Stage a contiguous 64KB block (global -> LDS) with CDNA5 async-to-LDS DMA.
// 256 threads x 16 iterations x 16B, coalesced. Tracked by ASYNCcnt.
static __device__ __forceinline__ void async_stage_64k(_Float16* lds, const _Float16* src, int tid) {
  unsigned lbase = (unsigned)(uintptr_t)(__attribute__((address_space(3))) _Float16*)lds;
  unsigned long long gbase = (unsigned long long)(uintptr_t)src;
  #pragma unroll
  for (int i = 0; i < 16; ++i) {
    unsigned off = (unsigned)((i * 256 + tid) * 16);
    unsigned la = lbase + off;
    unsigned long long ga = gbase + off;
    asm volatile("global_load_async_to_lds_b128 %0, %1, off" :: "v"(la), "v"(ga) : "memory");
  }
  asm volatile("s_wait_asynccnt 0x0" ::: "memory");
}

// ---------------------------------------------------------------- time MLP
__global__ void compute_t(const float* __restrict__ te,
                          const float* __restrict__ w1, const float* __restrict__ b1,
                          const float* __restrict__ w2, const float* __restrict__ b2,
                          float* __restrict__ tout) {
  __shared__ float tes[SIN_D];
  __shared__ float h1[T_DIM];
  int tid = threadIdx.x;
  if (tid < SIN_D) tes[tid] = te[tid];
  __syncthreads();
  float a = b1[tid];
  for (int i = 0; i < SIN_D; ++i) a += tes[i] * w1[i*T_DIM + tid];
  h1[tid] = a / (1.0f + __expf(-a));     // silu
  __syncthreads();
  float o = b2[tid];
  for (int i = 0; i < T_DIM; ++i) o += h1[i] * w2[i*T_DIM + tid];
  tout[tid] = o;
}

// ------------------------------------------- weight gen: W^T (f16) = t . src + bias
// src: [T_DIM][R][C] f32 ; bias: [R][C] ; dst: [C][R] f16 (transposed for WMMA-B)
__global__ void gen_weight_t(const float* __restrict__ src, const float* __restrict__ bias,
                             const float* __restrict__ tvec, _Float16* __restrict__ dst,
                             int R, int C) {
  __shared__ float ts[T_DIM];
  int tid = threadIdx.x;
  if (tid < T_DIM) ts[tid] = tvec[tid];
  __syncthreads();
  size_t idx = (size_t)blockIdx.x * blockDim.x + tid;
  int cq = C >> 2;
  if (idx >= (size_t)R * cq) return;
  int r = (int)(idx / cq);
  int c = (int)(idx % cq) * 4;
  const v4f* s4 = (const v4f*)(src + (size_t)r * C + c);
  size_t strideT = (size_t)R * C / 4;
  v4f acc = {};
  #pragma unroll 4
  for (int t = 0; t < T_DIM; ++t) acc += ts[t] * s4[(size_t)t * strideT];
  acc += *(const v4f*)(bias + (size_t)r * C + c);
  dst[(size_t)(c+0)*R + r] = (_Float16)acc[0];
  dst[(size_t)(c+1)*R + r] = (_Float16)acc[1];
  dst[(size_t)(c+2)*R + r] = (_Float16)acc[2];
  dst[(size_t)(c+3)*R + r] = (_Float16)acc[3];
}

// ---------------------------------------------- bias gen: dst[n] = sum_t t[t]*src[t][n]
__global__ void gen_bias(const float* __restrict__ src, const float* __restrict__ tvec,
                         float* __restrict__ dst, int N) {
  __shared__ float ts[T_DIM];
  int tid = threadIdx.x;
  if (tid < T_DIM) ts[tid] = tvec[tid];
  __syncthreads();
  int n = blockIdx.x * blockDim.x + tid;
  if (n >= N) return;
  float acc = 0.f;
  for (int t = 0; t < T_DIM; ++t) acc += ts[t] * src[(size_t)t * N + n];
  dst[n] = acc;
}

// ----------------------------------------------------- x: f32 -> f16 (one pass)
__global__ __launch_bounds__(256) void convert_x(const float* __restrict__ x,
                                                 _Float16* __restrict__ xh) {
  size_t i = ((size_t)blockIdx.x * blockDim.x + threadIdx.x) * 8;
  v4f a = *(const v4f*)(x + i);
  v4f b = *(const v4f*)(x + i + 4);
  v8h o;
  #pragma unroll
  for (int j = 0; j < 4; ++j) { o[j] = (_Float16)a[j]; o[4+j] = (_Float16)b[j]; }
  *(v8h*)(xh + i) = o;
}

// ------------------------------------------------------------- QKV GEMM (WMMA f16)
// Block owns one 64-col N-strip: B-strip (64x512 f16 = 64KB) async-staged to LDS once,
// 8 waves (8 M-tiles) consume it. C[m][c] = xh[m][:] @ W_attn[:][c] + b_attn[c].
__global__ __launch_bounds__(256) void qkv_gemm(const _Float16* __restrict__ xh,
                                                const _Float16* __restrict__ wta,   // [1536][512]
                                                const float* __restrict__ ba,
                                                _Float16* __restrict__ qb,
                                                _Float16* __restrict__ kbuf,
                                                _Float16* __restrict__ vtb) {
  __shared__ __align__(16) _Float16 bs[64 * EMBED];   // 64 KB
  int tid = threadIdx.x;
  int lane = tid & 31, wv = tid >> 5;
  int nstrip = blockIdx.x % (QKVC/64);
  int mb     = blockIdx.x / (QKVC/64);
  int n0 = nstrip * 64;
  int m0 = (mb * 8 + wv) * 16;

  async_stage_64k(bs, wta + (size_t)n0 * EMBED, tid);
  __syncthreads();

  int l15 = lane & 15; bool hi = lane >= 16;
  const _Float16* ar = xh + (size_t)(m0 + l15) * EMBED;
  v8f acc[4] = {};
  for (int k0 = 0; k0 < EMBED; k0 += 32) {
    if (k0 + 64 < EMBED) __builtin_prefetch(ar + k0 + 64, 0, 3);
    int ka = k0 + (hi ? 16 : 0);
    v16h a = cat16(ld8(ar + ka), ld8(ar + ka + 8));
    int kw = k0 + (hi ? 16 : 0);
    #pragma unroll
    for (int j = 0; j < 4; ++j) {
      const _Float16* wp = &bs[(j*16 + l15) * EMBED + kw];   // ds_load_b128 x2
      v16h bf = cat16(ld8(wp), ld8(wp + 8));
      acc[j] = WMMA(a, bf, acc[j]);
    }
  }
  #pragma unroll
  for (int j = 0; j < 4; ++j) {
    int c = n0 + j*16 + l15;
    int q3 = c >> 9, rem = c & 511;
    int h = rem >> 6, d = rem & 63;
    float bias = ba[c];
    #pragma unroll
    for (int r = 0; r < 8; ++r) {
      int row = m0 + r + (hi ? 8 : 0);
      float val = acc[j][r] + bias;
      int bb = row >> 10, p = row & (SEQ-1);
      size_t bhp = (size_t)(bb*HEADS + h) * SEQ + p;
      if (q3 == 0)      qb  [bhp*HDIM + d] = (_Float16)(val * 0.125f);   // 1/sqrt(64)
      else if (q3 == 1) kbuf[bhp*HDIM + d] = (_Float16)val;
      else              vtb [((size_t)(bb*HEADS + h)*HDIM + d)*SEQ + p] = (_Float16)val;
    }
  }
}

// ------------------------------------------------------ causal flash attention
__global__ __launch_bounds__(256) void attn_kernel(const _Float16* __restrict__ qb,
                                                   const _Float16* __restrict__ kb,
                                                   const _Float16* __restrict__ vtb,
                                                   _Float16* __restrict__ ob) {
  __shared__ __align__(16) _Float16 pls[8][16*32];
  int lane = threadIdx.x & 31, wv = threadIdx.x >> 5;
  int bh = blockIdx.x >> 3, qblk = blockIdx.x & 7;
  int b = bh >> 3, h = bh & 7;
  int q0 = (qblk * 8 + wv) * 16;
  const _Float16* qhp = qb  + (size_t)bh * (SEQ*HDIM);
  const _Float16* khp = kb  + (size_t)bh * (SEQ*HDIM);
  const _Float16* vhp = vtb + (size_t)bh * (SEQ*HDIM);
  _Float16* pl = &pls[wv][0];
  int l15 = lane & 15; bool hi = lane >= 16;
  int da = hi ? 16 : 0, db = hi ? 16 : 0;

  const _Float16* qr = qhp + (size_t)(q0 + l15) * HDIM;
  v16h a0 = cat16(ld8(qr + da),      ld8(qr + da + 8));
  v16h a1 = cat16(ld8(qr + 32 + da), ld8(qr + 32 + da + 8));

  v8f oacc[4] = {};
  float mrun[8], lrun[8];
  #pragma unroll
  for (int r = 0; r < 8; ++r) { mrun[r] = -1e30f; lrun[r] = 0.f; }
  int rowbase = q0 + (hi ? 8 : 0);

  for (int n0 = 0; n0 < q0 + 16; n0 += 32) {
    v8f s0 = {}, s1 = {};
    {
      const _Float16* kr = khp + (size_t)(n0 + l15) * HDIM;
      v16h b0 = cat16(ld8(kr + db),      ld8(kr + db + 8));
      v16h b1 = cat16(ld8(kr + 32 + db), ld8(kr + 32 + db + 8));
      s0 = WMMA(a0, b0, s0); s0 = WMMA(a1, b1, s0);
    }
    {
      const _Float16* kr = khp + (size_t)(n0 + 16 + l15) * HDIM;
      v16h b0 = cat16(ld8(kr + db),      ld8(kr + db + 8));
      v16h b1 = cat16(ld8(kr + 32 + db), ld8(kr + 32 + db + 8));
      s1 = WMMA(a0, b0, s1); s1 = WMMA(a1, b1, s1);
    }
    int key0 = n0 + l15, key1 = key0 + 16;
    #pragma unroll
    for (int r = 0; r < 8; ++r) {
      int row = rowbase + r;
      float v0 = (key0 <= row) ? s0[r] : -1e30f;
      float v1 = (key1 <= row) ? s1[r] : -1e30f;
      float mx = fmaxf(v0, v1);                    // row reduce across 16-lane half
      mx = fmaxf(mx, __shfl_xor(mx, 1)); mx = fmaxf(mx, __shfl_xor(mx, 2));
      mx = fmaxf(mx, __shfl_xor(mx, 4)); mx = fmaxf(mx, __shfl_xor(mx, 8));
      float mnew = fmaxf(mrun[r], mx);
      float p0 = __expf(v0 - mnew), p1 = __expf(v1 - mnew);
      float sum = p0 + p1;
      sum += __shfl_xor(sum, 1); sum += __shfl_xor(sum, 2);
      sum += __shfl_xor(sum, 4); sum += __shfl_xor(sum, 8);
      float corr = __expf(mrun[r] - mnew);
      lrun[r] = lrun[r] * corr + sum;
      mrun[r] = mnew;
      #pragma unroll
      for (int j = 0; j < 4; ++j) oacc[j][r] = oacc[j][r] * corr;
      int rl = (hi ? 8 : 0) + r;                   // C-layout -> row-major P in LDS
      pl[rl*32 + l15]      = (_Float16)p0;
      pl[rl*32 + 16 + l15] = (_Float16)p1;
    }
    asm volatile("s_wait_dscnt 0x0" ::: "memory"); // LDS round-trip fence (same wave)
    const _Float16* pr = pl + l15*32 + (hi ? 16 : 0);
    v16h pa = cat16(ld8(pr), ld8(pr + 8));         // P as A-fragment (16x32)
    #pragma unroll
    for (int j = 0; j < 4; ++j) {
      const _Float16* vr = vhp + ((size_t)(j*16 + l15) << 10) + n0 + db;
      v16h bv = cat16(ld8(vr), ld8(vr + 8));
      oacc[j] = WMMA(pa, bv, oacc[j]);
    }
  }
  #pragma unroll
  for (int j = 0; j < 4; ++j) {
    int d = j*16 + l15;
    #pragma unroll
    for (int r = 0; r < 8; ++r) {
      int p = rowbase + r;
      float val = oacc[j][r] / lrun[r];
      ob[(size_t)(b*SEQ + p) * EMBED + h*HDIM + d] = (_Float16)val;
    }
  }
}

// --------------------------------------------------------- output projection
// Same LDS-staged B-strip structure as qkv_gemm.
__global__ __launch_bounds__(256) void out_proj(const _Float16* __restrict__ ob,
                                                const _Float16* __restrict__ wpt,  // [512][512]
                                                const float* __restrict__ bp,
                                                float* __restrict__ out) {
  __shared__ __align__(16) _Float16 bs[64 * EMBED];   // 64 KB
  int tid = threadIdx.x;
  int lane = tid & 31, wv = tid >> 5;
  int nstrip = blockIdx.x & 7;
  int mb     = blockIdx.x >> 3;
  int n0 = nstrip * 64;
  int m0 = (mb * 8 + wv) * 16;

  async_stage_64k(bs, wpt + (size_t)n0 * EMBED, tid);
  __syncthreads();

  int l15 = lane & 15; bool hi = lane >= 16;
  const _Float16* ar = ob + (size_t)(m0 + l15) * EMBED;
  v8f acc[4] = {};
  for (int k0 = 0; k0 < EMBED; k0 += 32) {
    int ka = k0 + (hi ? 16 : 0);
    v16h a = cat16(ld8(ar + ka), ld8(ar + ka + 8));
    int kw = k0 + (hi ? 16 : 0);
    #pragma unroll
    for (int j = 0; j < 4; ++j) {
      const _Float16* wp = &bs[(j*16 + l15) * EMBED + kw];
      v16h bf = cat16(ld8(wp), ld8(wp + 8));
      acc[j] = WMMA(a, bf, acc[j]);
    }
  }
  #pragma unroll
  for (int j = 0; j < 4; ++j) {
    int c = n0 + j*16 + l15;
    float bias = bp[c];
    #pragma unroll
    for (int r = 0; r < 8; ++r) {
      int row = m0 + r + (hi ? 8 : 0);
      out[(size_t)row * EMBED + c] = acc[j][r] + bias;
    }
  }
}

// ---------------------------------------------------------------- launcher
extern "C" void kernel_launch(void* const* d_in, const int* in_sizes, int n_in,
                              void* d_out, int out_size, void* d_ws, size_t ws_size,
                              hipStream_t stream) {
  (void)in_sizes; (void)n_in; (void)out_size; (void)ws_size;
  const float* time_embed = (const float*)d_in[0];
  const float* x          = (const float*)d_in[1];
  const float* lin1_w     = (const float*)d_in[2];
  const float* lin1_b     = (const float*)d_in[3];
  const float* lin2_w     = (const float*)d_in[4];
  const float* lin2_b     = (const float*)d_in[5];
  const float* fW_attn_w  = (const float*)d_in[6];
  const float* fW_attn_b  = (const float*)d_in[7];
  const float* fb_attn    = (const float*)d_in[8];
  const float* fW_proj_w  = (const float*)d_in[9];
  const float* fW_proj_b  = (const float*)d_in[10];
  const float* fb_proj    = (const float*)d_in[11];
  float* out = (float*)d_out;

  char* ws = (char*)d_ws;
  float*     tvec = (float*)(ws + 0);                    // 128 f32
  float*     bA   = (float*)(ws + 1024);                 // 1536 f32
  float*     bP   = (float*)(ws + 8192);                 // 512 f32
  _Float16*  wta  = (_Float16*)(ws + 16384);             // 1536x512 f16 (W_attn^T)
  _Float16*  wpt  = (_Float16*)(ws + (2ull << 20));      // 512x512 f16  (W_proj^T)
  _Float16*  qb   = (_Float16*)(ws + (4ull  << 20));     // 16 MB  [b][h][p][d]
  _Float16*  kb   = (_Float16*)(ws + (20ull << 20));     // 16 MB  [b][h][p][d]
  _Float16*  vtb  = (_Float16*)(ws + (36ull << 20));     // 16 MB  [b][h][d][p]
  // xh and ob share one 16MB region: xh dies after qkv_gemm, ob is born in attn_kernel.
  _Float16*  xh   = (_Float16*)(ws + (52ull << 20));     // 16 MB  x as f16
  _Float16*  ob   = (_Float16*)(ws + (52ull << 20));     // 16 MB  [b][p][h*d]

  compute_t<<<1, 128, 0, stream>>>(time_embed, lin1_w, lin1_b, lin2_w, lin2_b, tvec);

  gen_weight_t<<<768, 256, 0, stream>>>(fW_attn_w, fW_attn_b, tvec, wta, EMBED, QKVC);
  gen_weight_t<<<256, 256, 0, stream>>>(fW_proj_w, fW_proj_b, tvec, wpt, EMBED, EMBED);
  gen_bias<<<6, 256, 0, stream>>>(fb_attn, tvec, bA, QKVC);
  gen_bias<<<2, 256, 0, stream>>>(fb_proj, tvec, bP, EMBED);

  convert_x<<<(MROWS*EMBED)/(8*256), 256, 0, stream>>>(x, xh);

  qkv_gemm<<<(MROWS/128) * (QKVC/64), 256, 0, stream>>>(xh, wta, bA, qb, kb, vtb);

  attn_kernel<<<BATCH * HEADS * (SEQ/128), 256, 0, stream>>>(qb, kb, vtb, ob);

  out_proj<<<(MROWS/128) * (EMBED/64), 256, 0, stream>>>(ob, wpt, bP, out);
}
